// _VGG16Head_23622320128428
// MI455X (gfx1250) — matrix operand order repaired and checked by reading, not compiled, not run on hardware
//
#include <hip/hip_runtime.h>
#include <hip/hip_bf16.h>

// ---------------------------------------------------------------------------
// VGG16 head for MI455X (gfx1250, wave32).
//   1) ROI max-pool (256 ROIs, 512ch, 7x7) -> bf16 activations in ws
//   2) bf16 WMMA GEMMs: fc1(25088->4096)+ReLU, fc2(4096->4096)+ReLU,
//      delta head (4096->84), score head (4096->21)
// Weights stream from HBM once as f32 (global_load_b128), converted to bf16
// on the fly; activations stage LDS via global_load_async_to_lds_b128
// (ASYNCcnt); math runs on v_wmma_f32_16x16x32_bf16 with f32 accumulation.
// ---------------------------------------------------------------------------

typedef __attribute__((ext_vector_type(16))) __bf16 v16bf;
typedef __attribute__((ext_vector_type(8)))  float  v8f;

union FragAB { unsigned u[8]; v16bf v; };

__device__ __forceinline__ unsigned short f2bf(float f) {
  unsigned u = __float_as_uint(f);
  u += 0x7FFFu + ((u >> 16) & 1u);   // round-to-nearest-even
  return (unsigned short)(u >> 16);
}

// ---------------------------------------------------------------------------
// ROI max pooling: one block per ROI, 256 threads sweep 512*49 bins.
// ---------------------------------------------------------------------------
__global__ __launch_bounds__(256)
void roi_pool_kernel(const float* __restrict__ feat,   // (512,50,50)
                     const float* __restrict__ rois,   // (256,4)
                     const int*   __restrict__ image_size,
                     unsigned short* __restrict__ Xbf) // (256, 25088) bf16
{
  const int C = 512, H = 50, W = 50, P = 7;
  const int n = blockIdx.x;
  const float scale = (float)H / (float)(*image_size);

  const float x1 = rois[n * 4 + 0], y1 = rois[n * 4 + 1];
  const float x2 = rois[n * 4 + 2], y2 = rois[n * 4 + 3];
  const int sw = (int)rintf(x1 * scale);
  const int sh = (int)rintf(y1 * scale);
  const int ew = (int)rintf(x2 * scale);
  const int eh = (int)rintf(y2 * scale);
  const float bw = (float)max(ew - sw + 1, 1) / (float)P;
  const float bh = (float)max(eh - sh + 1, 1) / (float)P;

  for (int idx = threadIdx.x; idx < C * P * P; idx += blockDim.x) {
    const int c = idx / (P * P);
    const int bin = idx % (P * P);
    const int ph = bin / P, pw = bin % P;
    const int hs = min(max((int)floorf((float)ph * bh) + sh, 0), H);
    const int he = min(max((int)ceilf((float)(ph + 1) * bh) + sh, 0), H);
    const int ws = min(max((int)floorf((float)pw * bw) + sw, 0), W);
    const int we = min(max((int)ceilf((float)(pw + 1) * bw) + sw, 0), W);
    float m = 0.0f;                                    // empty bin -> 0
    if (hs < he && ws < we) {
      m = -1e30f;
      const float* fp = feat + (size_t)c * H * W;
      for (int h = hs; h < he; ++h)
        for (int w = ws; w < we; ++w)
          m = fmaxf(m, fp[h * W + w]);
    }
    Xbf[(size_t)n * (C * P * P) + idx] = f2bf(m);
  }
}

// ---------------------------------------------------------------------------
// GEMM: C(M=256 x N) = relu?(A_bf16(256 x K) @ B_f32(K x N) + bias)
// 256 threads = 8 waves; wave w owns M rows [32w, 32w+32); block owns one
// 32-wide N stripe -> each weight element read from HBM exactly once.
// K-step = 32 (bf16 WMMA K). Each wave: 2x2 tiles of v_wmma_f32_16x16x32_bf16.
// BOUNDED=false (fc1/fc2): unguarded b128 weight loads, no exec divergence.
// ---------------------------------------------------------------------------
template <bool RELU, bool OUTBF16, bool BOUNDED>
__global__ __launch_bounds__(256)
void gemm_bf16_kernel(const unsigned short* __restrict__ A, // M x K (bf16 bits)
                      const float* __restrict__ B,          // K x N (f32)
                      const float* __restrict__ bias,       // N
                      void* __restrict__ Cout,
                      int N, int K, int ldc)
{
  // Inner stride 40 bf16 (80B) keeps b128 transfers 16B-aligned, staggers banks.
  __shared__ unsigned short sA[256 * 40]; // 256 x 32 bf16 A tile
  __shared__ unsigned short sB[32 * 40];  // transposed [n][k] 32x32 bf16 B tile

  const int tid   = threadIdx.x;
  const int lane  = tid & 31;
  const int waveM = tid >> 5;      // 0..7 -> rows [32*waveM, +32)
  const int g     = lane >> 4;     // half-wave: K-half selector
  const int ln    = lane & 15;     // row (A) / col (B) within 16
  const int nWG   = blockIdx.x * 32;

  v8f acc[2][2];
  #pragma unroll
  for (int i = 0; i < 2; ++i)
    #pragma unroll
    for (int j = 0; j < 2; ++j)
      #pragma unroll
      for (int r = 0; r < 8; ++r) acc[i][j][r] = 0.0f;

  // Per-thread fixed coordinates for the A async copy (4 x b128 per K-step).
  unsigned aLds[4];
  const unsigned short* aGlb[4];
  #pragma unroll
  for (int it = 0; it < 4; ++it) {
    const int vi  = tid + it * 256;          // 0..1023
    const int row = vi >> 2;                 // 0..255
    const int kc  = (vi & 3) * 8;            // 0,8,16,24
    aLds[it] = (unsigned)(size_t)&sA[row * 40 + kc];  // LDS byte offset
    aGlb[it] = A + (size_t)row * K + kc;
  }
  // Per-thread fixed coordinates for the B tile.
  const int bKr  = tid >> 3;                 // 0..31 (unbounded: float4/thread)
  const int bNc4 = (tid & 7) * 4;            // 0,4,...,28

  for (int kk = 0; kk < K; kk += 32) {
    // --- stage A tile: async DMA global -> LDS (ASYNCcnt), no VGPR bounce ---
    #pragma unroll
    for (int it = 0; it < 4; ++it) {
      const unsigned short* ga = aGlb[it] + kk;
      asm volatile("global_load_async_to_lds_b128 %0, %1, off"
                   :: "v"(aLds[it]), "v"(ga)
                   : "memory");
    }

    // --- stage B tile: 32x32 f32 -> bf16, transposed sB[n][k] ---
    if (!BOUNDED) {
      const float4 bv = *(const float4*)&B[(size_t)(kk + bKr) * N + nWG + bNc4];
      sB[(bNc4 + 0) * 40 + bKr] = f2bf(bv.x);
      sB[(bNc4 + 1) * 40 + bKr] = f2bf(bv.y);
      sB[(bNc4 + 2) * 40 + bKr] = f2bf(bv.z);
      sB[(bNc4 + 3) * 40 + bKr] = f2bf(bv.w);
      if (kk + 32 < K)  // prefetch next weight K-block
        __builtin_prefetch(&B[(size_t)(kk + 32 + bKr) * N + nWG + bNc4], 0, 1);
    } else {
      #pragma unroll
      for (int it = 0; it < 4; ++it) {
        const int e  = tid + it * 256;       // 0..1023
        const int kr = e >> 5;               // 0..31
        const int nc = e & 31;
        float v = 0.0f;
        if (nWG + nc < N) v = B[(size_t)(kk + kr) * N + nWG + nc];
        sB[nc * 40 + kr] = f2bf(v);
      }
    }

    // Own async copies must land before signaling; then barrier publishes LDS.
    asm volatile("s_wait_asynccnt 0x0" ::: "memory");
    __syncthreads();

    // --- build fragments per CDNA5 ISA 16-bit layouts ---
    FragAB fa[2], fb[2];
    #pragma unroll
    for (int i = 0; i < 2; ++i) {
      const int row = waveM * 32 + i * 16 + ln;   // A: lane holds row, K pairs
      #pragma unroll
      for (int q = 0; q < 8; ++q) {
        const int k0 = (q >> 2) * 16 + g * 8 + (q & 3) * 2;
        fa[i].u[q] = *(const unsigned*)&sA[row * 40 + k0];
      }
    }
    #pragma unroll
    for (int j = 0; j < 2; ++j) {
      const int col = j * 16 + ln;                // B: lane holds col, K pairs
      #pragma unroll
      for (int q = 0; q < 8; ++q) {
        const int k0 = (q >> 2) * 16 + g * 8 + (q & 3) * 2;
        fb[j].u[q] = *(const unsigned*)&sB[col * 40 + k0];
      }
    }

    #pragma unroll
    for (int i = 0; i < 2; ++i)
      #pragma unroll
      for (int j = 0; j < 2; ++j)
        acc[i][j] = __builtin_amdgcn_wmma_f32_16x16x32_bf16(
            false, fa[i].v, false, fb[j].v, (short)0, acc[i][j], false, false);
    __syncthreads();
  }

  // --- epilogue: bias (+ ReLU), store bf16 (intermediate) or f32 (heads) ---
  #pragma unroll
  for (int j = 0; j < 2; ++j) {
    const int col = nWG + j * 16 + ln;
    const float bv = (!BOUNDED || col < N) ? bias[col] : 0.0f;
    #pragma unroll
    for (int i = 0; i < 2; ++i) {
      #pragma unroll
      for (int r = 0; r < 8; ++r) {
        const int rowg = waveM * 32 + i * 16 + g * 8 + r; // C layout: VGPR r
        float v = acc[i][j][r] + bv;
        if (RELU) v = fmaxf(v, 0.0f);
        if (!BOUNDED || col < N) {
          if (OUTBF16)
            ((unsigned short*)Cout)[(size_t)rowg * ldc + col] = f2bf(v);
          else
            ((float*)Cout)[(size_t)rowg * ldc + col] = v;
        }
      }
    }
  }
}

// ---------------------------------------------------------------------------
extern "C" void kernel_launch(void* const* d_in, const int* in_sizes, int n_in,
                              void* d_out, int out_size, void* d_ws, size_t ws_size,
                              hipStream_t stream) {
  const float* feat = (const float*)d_in[0];
  const float* rois = (const float*)d_in[1];
  const float* W1   = (const float*)d_in[2];
  const float* b1   = (const float*)d_in[3];
  const float* W2   = (const float*)d_in[4];
  const float* b2   = (const float*)d_in[5];
  const float* Wd   = (const float*)d_in[6];
  const float* bd   = (const float*)d_in[7];
  const float* Wsc  = (const float*)d_in[8];
  const float* bs   = (const float*)d_in[9];
  const int*   img  = (const int*)d_in[10];

  const int M = 256, DIN = 25088, DH = 4096, ND = 84, NS = 21;

  unsigned short* X  = (unsigned short*)d_ws;        // 256 x 25088 bf16
  unsigned short* H1 = X  + (size_t)M * DIN;         // 256 x 4096  bf16
  unsigned short* H2 = H1 + (size_t)M * DH;          // 256 x 4096  bf16
  float* outD = (float*)d_out;                       // 256 x 84
  float* outS = outD + (size_t)M * ND;               // 256 x 21

  roi_pool_kernel<<<M, 256, 0, stream>>>(feat, rois, img, X);

  gemm_bf16_kernel<true,  true,  false><<<dim3(DH / 32), 256, 0, stream>>>(
      X,  W1, b1, H1, DH, DIN, DH);
  gemm_bf16_kernel<true,  true,  false><<<dim3(DH / 32), 256, 0, stream>>>(
      H1, W2, b2, H2, DH, DH, DH);
  gemm_bf16_kernel<false, false, true ><<<dim3((ND + 31) / 32), 256, 0, stream>>>(
      H2, Wd, bd, outD, ND, DH, ND);
  gemm_bf16_kernel<false, false, true ><<<dim3((NS + 31) / 32), 256, 0, stream>>>(
      H2, Wsc, bs, outS, NS, DH, NS);
}
